// PolicyNetwork_47545287966964
// MI455X (gfx1250) — matrix operand order repaired and checked by reading, not compiled, not run on hardware
//
#include <hip/hip_runtime.h>

// ---------------- problem constants ----------------
#define H     2048
#define T_SEQ 8192
#define G4H   8192          // 4*H
#define NBLK  64            // persistent blocks (512 waves total)
#define NTHR  256           // 8 waves of 32

typedef __bf16 bf16_t;
typedef __attribute__((ext_vector_type(16))) __bf16 v16bf;
typedef __attribute__((ext_vector_type(8)))  float  v8f;

// ---------------- workspace layout (bytes) ----------------
// packedW : bf16[8192*2048] swizzled for WMMA A-tiles      = 33,554,432
// bsum    : float[8192]  (b_ih + b_hh)                     =     32,768
// hbuf    : bf16[2][2048] double-buffered hidden state     =      8,192
// feat    : float[8192]  concat of 4 action h's            =     32,768
// hid     : float[2048]  MLP hidden                        =      8,192
// bar     : unsigned[2]  global barrier {count, gen}
#define OFF_PACKW 0ull
#define OFF_BSUM  33554432ull
#define OFF_HBUF  33587200ull
#define OFF_FEAT  33595392ull
#define OFF_HID   33628160ull
#define OFF_BAR   33636352ull

// ---------------- helpers ----------------
__device__ __forceinline__ float sigf(float z) { return 1.0f / (1.0f + __expf(-z)); }
__device__ __forceinline__ float tanhfast(float z) {
    z = fminf(15.0f, fmaxf(-15.0f, z));
    float e = __expf(2.0f * z);
    return (e - 1.0f) / (e + 1.0f);
}

// device-wide sense-reversing barrier (64 co-resident blocks)
__device__ __forceinline__ void gbar(unsigned* cnt, unsigned* gen) {
    __threadfence();
    __syncthreads();
    if (threadIdx.x == 0) {
        unsigned g = __hip_atomic_load(gen, __ATOMIC_RELAXED, __HIP_MEMORY_SCOPE_AGENT);
        unsigned a = __hip_atomic_fetch_add(cnt, 1u, __ATOMIC_ACQ_REL, __HIP_MEMORY_SCOPE_AGENT);
        if (a == NBLK - 1u) {
            __hip_atomic_store(cnt, 0u, __ATOMIC_RELAXED, __HIP_MEMORY_SCOPE_AGENT);
            __hip_atomic_fetch_add(gen, 1u, __ATOMIC_RELEASE, __HIP_MEMORY_SCOPE_AGENT);
        } else {
            while (__hip_atomic_load(gen, __ATOMIC_ACQUIRE, __HIP_MEMORY_SCOPE_AGENT) == g)
                __builtin_amdgcn_s_sleep(1);
        }
    }
    __syncthreads();
}

// One 16-row slice of gates = W_hh_tile(16 x 2048) . h  via 64 x WMMA bf16.
// A is pre-swizzled; B broadcasts the h chunk across all 16 columns, so every
// column of D carries the dot product.  Lane 0 holds rows 0..7, lane 16 rows 8..15.
__device__ __forceinline__ void gemv_step(const v16bf* __restrict__ packedW,
                                          const bf16_t* __restrict__ hcur,
                                          float* gbuf, int tileR, int w, int lane) {
    const v16bf* Ap = packedW + ((size_t)tileR * 64u * 32u + (unsigned)lane);
    const int koff = (lane & 16) ? 16 : 0;
    v8f acc = {0.f, 0.f, 0.f, 0.f, 0.f, 0.f, 0.f, 0.f};
#pragma unroll 4
    for (int kt = 0; kt < 64; ++kt) {
        v16bf a = Ap[kt * 32];                                   // 32B contiguous per lane
        v16bf b = *(const v16bf*)(hcur + kt * 32 + koff);        // broadcast h chunk
        acc = __builtin_amdgcn_wmma_f32_16x16x32_bf16(
            false, a, false, b, (short)0, acc, false, false);
    }
    if (lane == 0) {
#pragma unroll
        for (int m = 0; m < 8; ++m) gbuf[w * 16 + m] = acc[m];
    } else if (lane == 16) {
#pragma unroll
        for (int m = 0; m < 8; ++m) gbuf[w * 16 + 8 + m] = acc[m];
    }
}

// ---------------- prep kernels ----------------
// Swizzle fp32 W_hh (8192x2048) -> bf16 WMMA A-tile layout.
// p = (((tileR*64 + kt)*32 + lane)*16 + e), e = 2r+hi
// lane<16 : row = tileR*16+lane,    K = kt*32 + e
// lane>=16: row = tileR*16+lane-16, K = kt*32 + 16 + e
__global__ __launch_bounds__(256) void pack_whh_kernel(const float* __restrict__ W_hh,
                                                       bf16_t* __restrict__ packedW) {
    unsigned p  = blockIdx.x * 256u + threadIdx.x;          // 16,777,216 total
    unsigned e  = p & 15u;
    unsigned L  = (p >> 4) & 31u;
    unsigned kt = (p >> 9) & 63u;
    unsigned tr = p >> 15;
    unsigned r  = e >> 1, hi = e & 1u;
    unsigned row = tr * 16u + (L & 15u);
    unsigned k   = kt * 32u + ((L & 16u) ? 16u : 0u) + e;
    packedW[p] = (bf16_t)W_hh[(size_t)row * H + k];
}

__global__ __launch_bounds__(256) void prep_small_kernel(const float* __restrict__ b_ih,
                                                         const float* __restrict__ b_hh,
                                                         float* __restrict__ bsum,
                                                         bf16_t* __restrict__ hbuf,
                                                         unsigned* __restrict__ bar) {
    unsigned i = blockIdx.x * 256u + threadIdx.x;
    if (i < G4H)  bsum[i] = b_ih[i] + b_hh[i];
    if (i < 2 * H) hbuf[i] = (bf16_t)0.0f;
    if (i < 2)     bar[i] = 0u;
}

// ---------------- persistent fused LSTM + head ----------------
__global__ __launch_bounds__(NTHR) void lstm_policy_kernel(
    const v16bf* __restrict__ packedW, const float* __restrict__ bsum,
    bf16_t* __restrict__ hbuf, float* __restrict__ feat, float* __restrict__ hid,
    unsigned* bar,
    const float* __restrict__ x, const float* __restrict__ W_ih,
    const float* __restrict__ last_dist, const float* __restrict__ last_heading,
    const float* __restrict__ W1, const float* __restrict__ b1,
    const float* __restrict__ W2, const float* __restrict__ b2,
    float* __restrict__ out) {
    __shared__ float gbuf[128];      // i/f/g/o partial gates for this block's 32 h-indices
    __shared__ float c_state[32];    // persistent cell state

    const int tid   = threadIdx.x;
    const int lane  = tid & 31;
    const int w     = tid >> 5;          // wave 0..7
    const int b     = blockIdx.x;        // 0..63
    const int q     = w >> 1;            // gate (i,f,g,o)
    const int s     = w & 1;             // 16-row half of the block's 32 rows
    const int tileR = q * 128 + b * 2 + s;   // 512 16-row tiles of the 8192 gate rows
    const int bbase = b * 32;                // this block's h-index base

    if (tid < 32) c_state[tid] = 0.0f;
    __syncthreads();

    // ---------- sequential scan ----------
    for (int t = 0; t < T_SEQ; ++t) {
        const bf16_t* hcur = hbuf + (size_t)(t & 1) * H;
        gemv_step(packedW, hcur, gbuf, tileR, w, lane);
        __syncthreads();
        if (tid < 32) {
            float x0 = x[t * 3 + 0], x1 = x[t * 3 + 1], x2 = x[t * 3 + 2];
            float gv[4];
#pragma unroll
            for (int qq = 0; qq < 4; ++qq) {
                int row = qq * H + bbase + tid;
                float g = gbuf[(qq * 2 + (tid >> 4)) * 16 + (tid & 15)];
                g += x0 * W_ih[row * 3 + 0] + x1 * W_ih[row * 3 + 1] + x2 * W_ih[row * 3 + 2]
                   + bsum[row];
                gv[qq] = g;
            }
            float i = sigf(gv[0]), f = sigf(gv[1]);
            float g = tanhfast(gv[2]), o = sigf(gv[3]);
            float cn = f * c_state[tid] + i * g;
            c_state[tid] = cn;
            hbuf[(size_t)((t + 1) & 1) * H + bbase + tid] = (bf16_t)(o * tanhfast(cn));
        }
        gbar(bar, bar + 1);
    }

    // ---------- 4 single-step action rollouts (share one more GEMV of W_hh.h) ----------
    {
        const bf16_t* hcur = hbuf + (size_t)(T_SEQ & 1) * H;
        gemv_step(packedW, hcur, gbuf, tileR, w, lane);
        __syncthreads();
        if (tid < 32) {
            float ld = last_dist[0], lh = last_heading[0];
            float c0 = c_state[tid];
            float hh[4], wi0[4], wi1[4], wi2[4], bs[4];
#pragma unroll
            for (int qq = 0; qq < 4; ++qq) {
                int row = qq * H + bbase + tid;
                hh[qq]  = gbuf[(qq * 2 + (tid >> 4)) * 16 + (tid & 15)];
                wi0[qq] = W_ih[row * 3 + 0];
                wi1[qq] = W_ih[row * 3 + 1];
                wi2[qq] = W_ih[row * 3 + 2];
                bs[qq]  = bsum[row];
            }
#pragma unroll
            for (int a = 0; a < 4; ++a) {
                float ga[4];
#pragma unroll
                for (int qq = 0; qq < 4; ++qq)
                    ga[qq] = hh[qq] + ld * wi0[qq] + lh * wi1[qq] + (float)a * wi2[qq] + bs[qq];
                float i = sigf(ga[0]), f = sigf(ga[1]);
                float g = tanhfast(ga[2]), o = sigf(ga[3]);
                float ca = f * c0 + i * g;
                feat[a * H + bbase + tid] = o * tanhfast(ca);
            }
        }
        gbar(bar, bar + 1);
    }

    // ---------- MLP layer 1: hid = relu(feat @ W1.T + b1), W1 (2048 x 8192) ----------
    {
        int wg = b * 8 + w;                   // 0..511 waves, 4 outputs each
#pragma unroll
        for (int jj = 0; jj < 4; ++jj) {
            int j = wg * 4 + jj;
            const float* wr = W1 + (size_t)j * G4H;
            float p = 0.0f;
            for (int k = lane; k < G4H; k += 32) p += feat[k] * wr[k];
#pragma unroll
            for (int m = 16; m >= 1; m >>= 1) p += __shfl_xor(p, m, 32);
            if (lane == 0) hid[j] = fmaxf(p + b1[j], 0.0f);
        }
        gbar(bar, bar + 1);
    }

    // ---------- output: out = hid @ W2.T + b2, W2 (4 x 2048) ----------
    if (b == 0 && w < 4) {
        const float* wr = W2 + w * H;
        float p = 0.0f;
        for (int j = lane; j < H; j += 32) p += hid[j] * wr[j];
#pragma unroll
        for (int m = 16; m >= 1; m >>= 1) p += __shfl_xor(p, m, 32);
        if (lane == 0) out[w] = p + b2[w];
    }
}

// ---------------- host launcher ----------------
extern "C" void kernel_launch(void* const* d_in, const int* in_sizes, int n_in,
                              void* d_out, int out_size, void* d_ws, size_t ws_size,
                              hipStream_t stream) {
    (void)in_sizes; (void)n_in; (void)out_size; (void)ws_size;
    const float* x            = (const float*)d_in[0];
    const float* last_dist    = (const float*)d_in[1];
    const float* last_heading = (const float*)d_in[2];
    const float* W_ih         = (const float*)d_in[3];
    const float* W_hh         = (const float*)d_in[4];
    const float* b_ih         = (const float*)d_in[5];
    const float* b_hh         = (const float*)d_in[6];
    const float* W1           = (const float*)d_in[7];
    const float* b1           = (const float*)d_in[8];
    const float* W2           = (const float*)d_in[9];
    const float* b2           = (const float*)d_in[10];
    float* out = (float*)d_out;

    char* ws = (char*)d_ws;
    bf16_t*   packedW = (bf16_t*)(ws + OFF_PACKW);
    float*    bsum    = (float*)(ws + OFF_BSUM);
    bf16_t*   hbuf    = (bf16_t*)(ws + OFF_HBUF);
    float*    feat    = (float*)(ws + OFF_FEAT);
    float*    hid     = (float*)(ws + OFF_HID);
    unsigned* bar     = (unsigned*)(ws + OFF_BAR);

    // one-time-per-launch prep: bf16 swizzle of W_hh + bias sum + state/barrier init
    pack_whh_kernel<<<(G4H * H) / 256, 256, 0, stream>>>(W_hh, packedW);
    prep_small_kernel<<<G4H / 256, 256, 0, stream>>>(b_ih, b_hh, bsum, hbuf, bar);

    // persistent fused LSTM scan + action rollout + MLP head
    lstm_policy_kernel<<<NBLK, NTHR, 0, stream>>>(
        (const v16bf*)packedW, bsum, hbuf, feat, hid, bar,
        x, W_ih, last_dist, last_heading, W1, b1, W2, b2, out);
}
